// GATLayer_58652073394471
// MI455X (gfx1250) — matrix operand order repaired and checked by reading, not compiled, not run on hardware
//
#include <hip/hip_runtime.h>
#include <hip/hip_bf16.h>

#define N_NODES 8192
#define D_IN    256
#define D_OUT   128
#define H_HEADS 2
#define DO2     256          // H * D_OUT, also columns of h / out
#define E_EDGES 262144
#define LRELU_ALPHA 0.2f

typedef __attribute__((ext_vector_type(16))) __bf16 v16bf;
typedef __attribute__((ext_vector_type(8)))  float  v8f;

// ---------------------------------------------------------------------------
// 1) Split f32 -> bf16 hi + bf16 lo (error-compensated operands for WMMA)
// ---------------------------------------------------------------------------
__global__ void cvt_split_kernel(const float* __restrict__ src,
                                 __bf16* __restrict__ hi,
                                 __bf16* __restrict__ lo, int n) {
    int i = blockIdx.x * blockDim.x + threadIdx.x;
    if (i < n) {
        float v  = src[i];
        __bf16 h = (__bf16)v;
        hi[i] = h;
        lo[i] = (__bf16)(v - (float)h);
    }
}

// ---------------------------------------------------------------------------
// 2) Projection GEMM: h[n][c] = sum_f x[n][f] * Wflat[c][f]
//    (Wflat = W viewed as (H*D_OUT, D_IN) row-major; c = head*128 + d)
//    One wave per 16x16 output tile; bf16x3 WMMA accumulation in f32.
// ---------------------------------------------------------------------------
__global__ void __launch_bounds__(256) gemm_wmma_kernel(
    const __bf16* __restrict__ xhi, const __bf16* __restrict__ xlo,
    const __bf16* __restrict__ whi, const __bf16* __restrict__ wlo,
    float* __restrict__ hOut) {
    const int wave = threadIdx.x >> 5;
    const int lane = threadIdx.x & 31;
    const int tile = blockIdx.x * 8 + wave;       // 512 * 16 = 8192 tiles
    const int m0 = (tile >> 4) << 4;              // node-row tile base
    const int n0 = (tile & 15) << 4;              // output-col tile base

    // A fragment addressing (16-bit A 16x32 layout): row = lane&15,
    // K chunks {kA..kA+7} and {kA+16..kA+23}, kA = 0 (lanes 0-15) / 8 (16-31)
    const int rowA = m0 + (lane & 15);
    const int kA   = (lane >> 4) * 8;
    // B fragment (32x16): col = lane&15, K chunks {kB..kB+7} and {kB+16..kB+23}, kB = 0 / 8
    const int colB = n0 + (lane & 15);
    const int kB   = (lane >> 4) * 8;

    v8f acc = {};
    for (int ks = 0; ks < D_IN; ks += 32) {
        union Frag { uint4 q[2]; v16bf v; } ah, al, bh, bl;
        const uint4* pxh = (const uint4*)(xhi + (size_t)rowA * D_IN + ks + kA);
        const uint4* pxl = (const uint4*)(xlo + (size_t)rowA * D_IN + ks + kA);
        ah.q[0] = pxh[0]; ah.q[1] = pxh[2];       // +16 bf16 = +2 uint4
        al.q[0] = pxl[0]; al.q[1] = pxl[2];
        const uint4* pwh = (const uint4*)(whi + (size_t)colB * D_IN + ks + kB);
        const uint4* pwl = (const uint4*)(wlo + (size_t)colB * D_IN + ks + kB);
        bh.q[0] = pwh[0]; bh.q[1] = pwh[2];
        bl.q[0] = pwl[0]; bl.q[1] = pwl[2];

        acc = __builtin_amdgcn_wmma_f32_16x16x32_bf16(false, ah.v, false, bh.v,
                                                      (short)0, acc, false, false);
        acc = __builtin_amdgcn_wmma_f32_16x16x32_bf16(false, ah.v, false, bl.v,
                                                      (short)0, acc, false, false);
        acc = __builtin_amdgcn_wmma_f32_16x16x32_bf16(false, al.v, false, bh.v,
                                                      (short)0, acc, false, false);
    }
    // C/D layout: lanes 0-15 -> M = r, lanes 16-31 -> M = 8 + r; N = lane&15
    const int rbase = m0 + (lane >> 4) * 8;
    const int col   = n0 + (lane & 15);
    for (int r = 0; r < 8; ++r)
        hOut[(size_t)(rbase + r) * DO2 + col] = acc[r];
}

// ---------------------------------------------------------------------------
// 3) Per-node attention logits: s_src[h][n] = h[n,h,:]·a[h,:128],
//                               s_dst[h][n] = h[n,h,:]·a[h,128:]
// ---------------------------------------------------------------------------
__global__ void scores_kernel(const float* __restrict__ h,
                              const float* __restrict__ a,
                              float* __restrict__ ssrc,
                              float* __restrict__ sdst) {
    int n = blockIdx.x * blockDim.x + threadIdx.x;
    if (n >= N_NODES) return;
    for (int hd = 0; hd < H_HEADS; ++hd) {
        const float* hv = h + (size_t)n * DO2 + hd * D_OUT;
        const float* as = a + hd * 2 * D_OUT;
        const float* ad = as + D_OUT;
        float s0 = 0.f, s1 = 0.f;
        for (int d = 0; d < D_OUT; ++d) { s0 += hv[d] * as[d]; s1 += hv[d] * ad[d]; }
        ssrc[hd * N_NODES + n] = s0;
        sdst[hd * N_NODES + n] = s1;
    }
}

// ---------------------------------------------------------------------------
// 4) Dense background term: sumh[c] = sum_n h[n][c]  (non-edges contribute
//    exp(0)=1 each -> numerator base is column sum, Z base is N)
// ---------------------------------------------------------------------------
__global__ void colsum_kernel(const float* __restrict__ h,
                              float* __restrict__ sumh) {
    __shared__ float red[256];
    int c = blockIdx.x;                              // 256 columns
    float s = 0.f;
    for (int n = threadIdx.x; n < N_NODES; n += blockDim.x)
        s += h[(size_t)n * DO2 + c];
    red[threadIdx.x] = s;
    __syncthreads();
    for (int off = 128; off > 0; off >>= 1) {
        if ((int)threadIdx.x < off) red[threadIdx.x] += red[threadIdx.x + off];
        __syncthreads();
    }
    if (threadIdx.x == 0) sumh[c] = red[0];
}

// ---------------------------------------------------------------------------
// 5) Edge kernel: one wave per edge. Dedup via NxN bit-matrix (set semantics
//    of adj). w = exp(leakyrelu(s_i + s_j)) - 1; scatter w*h[j,:] into acc[i,:]
//    and w into Zextra[i]. Duplicates compute identical values, so the
//    atomicOr first-setter race is deterministic in effect.
// ---------------------------------------------------------------------------
__global__ void __launch_bounds__(256) edge_kernel(
    const int* __restrict__ ei, const float* __restrict__ h,
    const float* __restrict__ ssrc, const float* __restrict__ sdst,
    unsigned* __restrict__ adjmask, float* __restrict__ zextra,
    float* __restrict__ acc) {
    const int wave = threadIdx.x >> 5;
    const int lane = threadIdx.x & 31;
    const int e = blockIdx.x * 8 + wave;
    if (e >= E_EDGES) return;
    const int i = ei[e];
    const int j = ei[E_EDGES + e];

    unsigned long long key = (unsigned long long)i * N_NODES + (unsigned)j;
    unsigned word = (unsigned)(key >> 5);
    unsigned bit  = 1u << (key & 31);
    unsigned old  = 0;
    if (lane == 0) old = atomicOr(&adjmask[word], bit);
    old = __shfl(old, 0, 32);
    if (old & bit) return;                          // duplicate edge -> drop

    for (int hd = 0; hd < H_HEADS; ++hd) {
        float sc = ssrc[hd * N_NODES + i] + sdst[hd * N_NODES + j];
        sc = sc > 0.f ? sc : LRELU_ALPHA * sc;
        float w = __expf(sc) - 1.f;
        if (lane == 0) atomicAdd(&zextra[hd * N_NODES + i], w);
        const float4 hv = *(const float4*)(h + (size_t)j * DO2 + hd * D_OUT + lane * 4);
        float* dst = acc + (size_t)i * DO2 + hd * D_OUT + lane * 4;
        atomicAdd(dst + 0, w * hv.x);
        atomicAdd(dst + 1, w * hv.y);
        atomicAdd(dst + 2, w * hv.z);
        atomicAdd(dst + 3, w * hv.w);
    }
}

// ---------------------------------------------------------------------------
// 6) Finalize: out[n][c] = ELU( (sumh[c] + acc[n][c]) / (N + Zextra[head][n]) )
// ---------------------------------------------------------------------------
__global__ void finalize_kernel(const float* __restrict__ acc,
                                const float* __restrict__ sumh,
                                const float* __restrict__ zextra,
                                float* __restrict__ out) {
    int idx = blockIdx.x * blockDim.x + threadIdx.x;
    if (idx >= N_NODES * DO2) return;
    int n  = idx >> 8;
    int c  = idx & 255;
    int hd = c >> 7;
    float num = sumh[c] + acc[idx];
    float Z   = (float)N_NODES + zextra[hd * N_NODES + n];
    float v   = num / Z;
    out[idx]  = v > 0.f ? v : (__expf(v) - 1.f);
}

// ---------------------------------------------------------------------------
extern "C" void kernel_launch(void* const* d_in, const int* in_sizes, int n_in,
                              void* d_out, int out_size, void* d_ws, size_t ws_size,
                              hipStream_t stream) {
    const float* x  = (const float*)d_in[0];   // (N, D_IN)
    const float* W  = (const float*)d_in[1];   // (H, D_OUT, D_IN) == (256, 256) flat
    const float* a  = (const float*)d_in[2];   // (H, 2*D_OUT)
    const int*   ei = (const int*)d_in[3];     // (2, E)
    float* out = (float*)d_out;

    char* ws = (char*)d_ws;
    size_t off = 0;
    auto alloc = [&](size_t bytes) -> char* {
        char* p = ws + off;
        off += (bytes + 255) & ~(size_t)255;
        return p;
    };
    __bf16* xhi = (__bf16*)alloc((size_t)N_NODES * D_IN * sizeof(__bf16));
    __bf16* xlo = (__bf16*)alloc((size_t)N_NODES * D_IN * sizeof(__bf16));
    __bf16* whi = (__bf16*)alloc((size_t)DO2 * D_IN * sizeof(__bf16));
    __bf16* wlo = (__bf16*)alloc((size_t)DO2 * D_IN * sizeof(__bf16));
    float* hbuf = (float*)alloc((size_t)N_NODES * DO2 * sizeof(float));
    float* ssrc = (float*)alloc((size_t)H_HEADS * N_NODES * sizeof(float));
    float* sdst = (float*)alloc((size_t)H_HEADS * N_NODES * sizeof(float));
    float* sumh = (float*)alloc((size_t)DO2 * sizeof(float));
    float* zext = (float*)alloc((size_t)H_HEADS * N_NODES * sizeof(float));
    float* acc  = (float*)alloc((size_t)N_NODES * DO2 * sizeof(float));
    unsigned* adjmask = (unsigned*)alloc((size_t)N_NODES * N_NODES / 8);   // 8 MB

    // Per-call zero init of accumulators & dedup mask (harness doesn't re-poison)
    hipMemsetAsync(acc, 0, (size_t)N_NODES * DO2 * sizeof(float), stream);
    hipMemsetAsync(zext, 0, (size_t)H_HEADS * N_NODES * sizeof(float), stream);
    hipMemsetAsync(adjmask, 0, (size_t)N_NODES * N_NODES / 8, stream);

    // 1) split operands to bf16 hi/lo
    {
        int n = N_NODES * D_IN;
        cvt_split_kernel<<<(n + 255) / 256, 256, 0, stream>>>(x, xhi, xlo, n);
        int m = DO2 * D_IN;
        cvt_split_kernel<<<(m + 255) / 256, 256, 0, stream>>>(W, whi, wlo, m);
    }
    // 2) projection GEMM via WMMA (8192 tiles / 8 waves per block)
    gemm_wmma_kernel<<<1024, 256, 0, stream>>>(xhi, xlo, whi, wlo, hbuf);
    // 3) per-node logits
    scores_kernel<<<N_NODES / 256, 256, 0, stream>>>(hbuf, a, ssrc, sdst);
    // 4) dense background column sums
    colsum_kernel<<<DO2, 256, 0, stream>>>(hbuf, sumh);
    // 5) sparse edge corrections (wave per edge)
    edge_kernel<<<E_EDGES / 8, 256, 0, stream>>>(ei, hbuf, ssrc, sdst,
                                                 adjmask, zext, acc);
    // 6) normalize + ELU
    finalize_kernel<<<(N_NODES * DO2) / 256, 256, 0, stream>>>(acc, sumh, zext, out);
    (void)in_sizes; (void)n_in; (void)out_size; (void)ws_size;
}
